// PointAttention_3435973837330
// MI455X (gfx1250) — matrix operand order repaired and checked by reading, not compiled, not run on hardware
//
#include <hip/hip_runtime.h>
#include <hip/hip_bf16.h>

// Shapes fixed by the reference.
#define BB 4
#define CC 256
#define NN 4096
#define CQd 64

typedef _Float16 v16h __attribute__((ext_vector_type(16)));
typedef _Float16 h8   __attribute__((ext_vector_type(8)));
typedef float    v8f  __attribute__((ext_vector_type(8)));

static __device__ inline v8f wmma16(v16h a, v16h b, v8f c) {
  // D = A(16x32 f16) * B(32x16 f16) + C(16x16 f32)
  return __builtin_amdgcn_wmma_f32_16x16x32_f16(false, a, false, b, (short)0, c,
                                                false, false);
}
static __device__ inline float bperm(int srcLane, float v) {
  return __int_as_float(
      __builtin_amdgcn_ds_bpermute(srcLane << 2, __float_as_int(v)));
}

// ---------------------------------------------------------------------------
// Kernel 1: pointwise projections q,k,v (fp32 math) -> f16 scratch.
// qT/kT stored [b][n][64] (channel-contiguous, matches WMMA A/B operand loads)
// v16  stored  [b][c][n]  (n-contiguous, matches WMMA A operand loads)
// grid = (BB*(NN/64), 6), block = (64,4). seg 0=q, 1=k, 2..5=v quarters.
// ---------------------------------------------------------------------------
__global__ __launch_bounds__(256) void proj_kernel(
    const float* __restrict__ x,
    const float* __restrict__ wq, const float* __restrict__ bq,
    const float* __restrict__ wk, const float* __restrict__ bk,
    const float* __restrict__ wv, const float* __restrict__ bv,
    _Float16* __restrict__ qT, _Float16* __restrict__ kT,
    _Float16* __restrict__ v16p) {
  int seg = blockIdx.y;
  int bn  = blockIdx.x;
  int bz  = bn / (NN / 64);
  int n   = (bn % (NN / 64)) * 64 + threadIdx.x;
  int obase = threadIdx.y * 16;

  const float* w; const float* bias; int wrow0;
  if (seg == 0)      { w = wq; bias = bq; wrow0 = 0; }
  else if (seg == 1) { w = wk; bias = bk; wrow0 = 0; }
  else               { w = wv; bias = bv; wrow0 = (seg - 2) * 64; }

  float acc[16];
#pragma unroll
  for (int j = 0; j < 16; j++) acc[j] = bias[wrow0 + obase + j];

  const float* xp = x + (size_t)bz * CC * NN + n;
  for (int c = 0; c < CC; c++) {
    float xv = xp[(size_t)c * NN];
    const float* wr = w + (size_t)(wrow0 + obase) * CC + c;
#pragma unroll
    for (int j = 0; j < 16; j++) acc[j] = fmaf(wr[(size_t)j * CC], xv, acc[j]);
  }

  if (seg < 2) {
    _Float16* dst = (seg == 0 ? qT : kT) + ((size_t)bz * NN + n) * CQd + obase;
#pragma unroll
    for (int j = 0; j < 16; j++) dst[j] = (_Float16)acc[j];
  } else {
#pragma unroll
    for (int j = 0; j < 16; j++) {
      int cc = wrow0 + obase + j;
      v16p[((size_t)bz * CC + cc) * NN + n] = (_Float16)acc[j];
    }
  }
}

// ---------------------------------------------------------------------------
// Kernel 2: row statistics. For each (b,n): mx = max_m s[n,m],
// inv = 1/sum_m exp(s[n,m]-mx). s computed with WMMA (2 x 16x16x32, K=CQ=64).
// One wave per 16-row n tile; online (max,sum) over 256 m tiles; 16-lane
// butterfly reduce across columns. stats[b*N+n] = {mx, inv}.
// ---------------------------------------------------------------------------
__global__ __launch_bounds__(256) void rowstats_kernel(
    const _Float16* __restrict__ qT, const _Float16* __restrict__ kT,
    float2* __restrict__ stats) {
  int wid  = blockIdx.x * 8 + (threadIdx.x >> 5);
  int lane = threadIdx.x & 31;
  int col  = lane & 15, half = lane >> 4, hl8 = half << 3;
  int ntile = wid & 255, bz = wid >> 8;

  // A operand: q rows n0..n0+15, K = channel o (two 32-wide chunks)
  int n = ntile * 16 + col;
  const h8* qp = (const h8*)(qT + ((size_t)bz * NN + n) * CQd);
  v16h a0, a1;
  {
    h8 t0 = qp[2 * half], t1 = qp[2 * half + 1], t2 = qp[4 + 2 * half], t3 = qp[5 + 2 * half];
#pragma unroll
    for (int i = 0; i < 8; i++) {
      a0[i] = t0[i]; a0[8 + i] = t1[i];
      a1[i] = t2[i]; a1[8 + i] = t3[i];
    }
  }

  float mx[8], sm[8];
#pragma unroll
  for (int r = 0; r < 8; r++) { mx[r] = -1e30f; sm[r] = 0.f; }

  for (int mt = 0; mt < NN / 16; mt++) {
    int m = mt * 16 + col;
    const v16h* kp = (const v16h*)(kT + ((size_t)bz * NN + m) * CQd);
    v16h b0 = kp[half], b1 = kp[2 + half];
    v8f acc = {};
    acc = wmma16(a0, b0, acc);
    acc = wmma16(a1, b1, acc);
#pragma unroll
    for (int r = 0; r < 8; r++) {
      float sv = acc[r];
      float nm = fmaxf(mx[r], sv);
      sm[r] = sm[r] * __expf(mx[r] - nm) + __expf(sv - nm);
      mx[r] = nm;
    }
  }
  // reduce across the 16 column lanes of each half-wave
#pragma unroll
  for (int mask = 1; mask < 16; mask <<= 1) {
#pragma unroll
    for (int r = 0; r < 8; r++) {
      float om = __shfl_xor(mx[r], mask, 32);
      float os = __shfl_xor(sm[r], mask, 32);
      float nm = fmaxf(mx[r], om);
      sm[r] = sm[r] * __expf(mx[r] - nm) + os * __expf(om - nm);
      mx[r] = nm;
    }
  }
  if (col == 0) {
    float2* sp = stats + (size_t)bz * NN + ntile * 16 + hl8;
#pragma unroll
    for (int r = 0; r < 8; r++) sp[r] = make_float2(mx[r], 1.0f / sm[r]);
  }
}

// ---------------------------------------------------------------------------
// Kernel 3: o[b,c,m] = sum_n v[c,n] * beta[n,m]; out = gamma*o + x.
// Wave owns (b, 16-wide m tile, 128-wide c half): per 32-wide n chunk:
// recompute s via 4 WMMAs (k B-operand is loop invariant), beta in
// accumulator layout, ds_bpermute to B-operand layout, 8 output WMMAs.
// 2048 waves total; 64 accumulator VGPRs/wave.
// ---------------------------------------------------------------------------
__global__ __launch_bounds__(256) void attn_out_kernel(
    const float* __restrict__ x, const _Float16* __restrict__ qT,
    const _Float16* __restrict__ kT, const _Float16* __restrict__ v16p,
    const float2* __restrict__ stats, const float* __restrict__ gamma,
    float* __restrict__ out) {
  int wid  = blockIdx.x * 8 + (threadIdx.x >> 5);
  int lane = threadIdx.x & 31;
  int col  = lane & 15, half = lane >> 4, hl8 = half << 3;
  int ch2 = wid & 1, mtile = (wid >> 1) & 255, bz = wid >> 9;
  int m0 = mtile * 16, c0 = ch2 * 128;

  // loop-invariant k B-operand for this m tile (K = channel o)
  int m = m0 + col;
  const v16h* kp = (const v16h*)(kT + ((size_t)bz * NN + m) * CQd);
  v16h bk0 = kp[half], bk1 = kp[2 + half];

  v8f accO[8];
#pragma unroll
  for (int t = 0; t < 8; t++) { v8f z = {}; accO[t] = z; }
  float g = gamma[0];

  for (int ch = 0; ch < NN / 32; ch++) {
    int nb = ch * 32;
    v8f s0 = {}, s1 = {};
    {  // s rows nb..nb+15
      const h8* qp = (const h8*)(qT + ((size_t)bz * NN + nb + col) * CQd);
      v16h a0, a1;
      h8 t0 = qp[2 * half], t1 = qp[2 * half + 1], t2 = qp[4 + 2 * half], t3 = qp[5 + 2 * half];
#pragma unroll
      for (int i = 0; i < 8; i++) {
        a0[i] = t0[i]; a0[8 + i] = t1[i];
        a1[i] = t2[i]; a1[8 + i] = t3[i];
      }
      s0 = wmma16(a0, bk0, s0);
      s0 = wmma16(a1, bk1, s0);
    }
    {  // s rows nb+16..nb+31
      const h8* qp = (const h8*)(qT + ((size_t)bz * NN + nb + 16 + col) * CQd);
      v16h a0, a1;
      h8 t0 = qp[2 * half], t1 = qp[2 * half + 1], t2 = qp[4 + 2 * half], t3 = qp[5 + 2 * half];
#pragma unroll
      for (int i = 0; i < 8; i++) {
        a0[i] = t0[i]; a0[8 + i] = t1[i];
        a1[i] = t2[i]; a1[8 + i] = t3[i];
      }
      s1 = wmma16(a0, bk0, s1);
      s1 = wmma16(a1, bk1, s1);
    }
    // beta in C/D accumulator layout: element r holds row n = nb(+16) + r + hl8
    const float2* sp0 = stats + (size_t)bz * NN + nb + hl8;
    const float2* sp1 = sp0 + 16;
    float beta0[8], beta1[8];
#pragma unroll
    for (int r = 0; r < 8; r++) {
      float2 st = sp0[r];
      beta0[r] = __expf(s0[r] - st.x) * st.y;
    }
#pragma unroll
    for (int r = 0; r < 8; r++) {
      float2 st = sp1[r];
      beta1[r] = __expf(s1[r] - st.x) * st.y;
    }
    // convert to B-operand layout: lane col = m-local, element e -> K row
    v16h bb;
#pragma unroll
    for (int e = 0; e < 16; e++) {
      int src = col + ((e < 8) ? 0 : 16);
      float f0 = bperm(src, beta0[e & 7]);
      float f1 = bperm(src, beta1[e & 7]);
      bb[e] = (_Float16)(half ? f1 : f0);
    }
    // accumulate o for 8 c tiles
#pragma unroll
    for (int t = 0; t < 8; t++) {
      const h8* vp =
          (const h8*)(v16p + ((size_t)bz * CC + c0 + t * 16 + col) * NN + nb);
      v16h av;
      h8 u0 = vp[2 * half], u1 = vp[2 * half + 1];
#pragma unroll
      for (int i = 0; i < 8; i++) { av[i] = u0[i]; av[8 + i] = u1[i]; }
      accO[t] = wmma16(av, bb, accO[t]);
    }
  }
  // epilogue: out = gamma*o + x
#pragma unroll
  for (int t = 0; t < 8; t++) {
#pragma unroll
    for (int r = 0; r < 8; r++) {
      int c = c0 + t * 16 + r + hl8;
      size_t idx = ((size_t)bz * CC + c) * NN + m0 + col;
      out[idx] = g * accO[t][r] + x[idx];
    }
  }
}

// ---------------------------------------------------------------------------
extern "C" void kernel_launch(void* const* d_in, const int* in_sizes, int n_in,
                              void* d_out, int out_size, void* d_ws,
                              size_t ws_size, hipStream_t stream) {
  const float* x     = (const float*)d_in[0];
  const float* wq    = (const float*)d_in[1];
  const float* bq    = (const float*)d_in[2];
  const float* wk    = (const float*)d_in[3];
  const float* bk    = (const float*)d_in[4];
  const float* wv    = (const float*)d_in[5];
  const float* bv    = (const float*)d_in[6];
  const float* gamma = (const float*)d_in[7];
  float* out = (float*)d_out;

  // workspace layout (all offsets 256B aligned): qT | kT | v16 | stats
  const size_t QT_BYTES = (size_t)BB * NN * CQd * sizeof(_Float16);  // 2 MB
  const size_t V_BYTES  = (size_t)BB * CC * NN * sizeof(_Float16);   // 8 MB
  char* ws = (char*)d_ws;
  _Float16* qT   = (_Float16*)ws;
  _Float16* kT   = (_Float16*)(ws + QT_BYTES);
  _Float16* v16p = (_Float16*)(ws + 2 * QT_BYTES);
  float2*   st   = (float2*)(ws + 2 * QT_BYTES + V_BYTES);

  proj_kernel<<<dim3(BB * (NN / 64), 6), dim3(64, 4), 0, stream>>>(
      x, wq, bq, wk, bk, wv, bv, qT, kT, v16p);

  // BB*(NN/16) = 1024 waves, 8 waves/block
  rowstats_kernel<<<128, 256, 0, stream>>>(qT, kT, st);

  // BB*(NN/16)*(CC/128) = 2048 waves, 8 waves/block
  attn_out_kernel<<<256, 256, 0, stream>>>(x, qT, kT, v16p, st, gamma, out);
}
